// pc_tokenizer_14860586844488
// MI455X (gfx1250) — matrix operand order repaired and checked by reading, not compiled, not run on hardware
//
#include <hip/hip_runtime.h>
#include <stdint.h>

#define BATCH 4
#define KNB   32
#define R2V   0.01f
#define EPSV  1e-5f
#define BIGV  1e10f

typedef __attribute__((ext_vector_type(16))) _Float16 v16h;
typedef __attribute__((ext_vector_type(8)))  float    v8f;

__device__ __forceinline__ uint32_t pack2h(float a, float b) {
  union { _Float16 h[2]; uint32_t u; } v;
  v.h[0] = (_Float16)a;
  v.h[1] = (_Float16)b;
  return v.u;
}

// ---------------------------------------------------------------------------
// Farthest point sampling: one block per batch. dists lives in LDS.
// ---------------------------------------------------------------------------
__global__ void fps_kernel(const float* __restrict__ p, int* __restrict__ idx,
                           int N, int m) {
  __shared__ float dist[8192];
  __shared__ float rmax[512];
  __shared__ int   rarg[512];
  __shared__ int   s_last;
  const int b = blockIdx.x;
  const int t = threadIdx.x;
  const int T = blockDim.x;
  const float* pb = p + (size_t)b * N * 3;
  for (int i = t; i < N; i += T) dist[i] = BIGV;
  if (t == 0) { idx[(size_t)b * m] = 0; s_last = 0; }
  __syncthreads();
  for (int it = 1; it < m; ++it) {
    const int last = s_last;
    const float lx = pb[last * 3 + 0];
    const float ly = pb[last * 3 + 1];
    const float lz = pb[last * 3 + 2];
    float bm = -1.0f; int ba = 0;
    for (int i = t; i < N; i += T) {
      float dx = pb[i * 3 + 0] - lx;
      float dy = pb[i * 3 + 1] - ly;
      float dz = pb[i * 3 + 2] - lz;
      float d  = dx * dx + dy * dy + dz * dz;
      float dd = fminf(dist[i], d);
      dist[i] = dd;
      if (dd > bm) { bm = dd; ba = i; }   // strided i ascending -> first max kept
    }
    rmax[t] = bm; rarg[t] = ba;
    __syncthreads();
    for (int s = T >> 1; s > 0; s >>= 1) {
      if (t < s) {
        float ov = rmax[t + s]; int oa = rarg[t + s];
        if (ov > rmax[t] || (ov == rmax[t] && oa < rarg[t])) { rmax[t] = ov; rarg[t] = oa; }
      }
      __syncthreads();
    }
    if (t == 0) { s_last = rarg[0]; idx[(size_t)b * m + it] = rarg[0]; }
    __syncthreads();
  }
}

// ---------------------------------------------------------------------------
// Gather sampled centers: cp (B,m,3), cf (B,C,m)
// ---------------------------------------------------------------------------
__global__ void gather_centers(const float* __restrict__ p, const float* __restrict__ f,
                               const int* __restrict__ idx, float* __restrict__ cp,
                               float* __restrict__ cf, int N, int m, int C) {
  const int b = blockIdx.y;
  const int i = blockIdx.x * blockDim.x + threadIdx.x;
  if (i >= m) return;
  const int id = idx[(size_t)b * m + i];
  cp[((size_t)b * m + i) * 3 + 0] = p[((size_t)b * N + id) * 3 + 0];
  cp[((size_t)b * m + i) * 3 + 1] = p[((size_t)b * N + id) * 3 + 1];
  cp[((size_t)b * m + i) * 3 + 2] = p[((size_t)b * N + id) * 3 + 2];
  for (int c = 0; c < C; ++c)
    cf[((size_t)b * C + c) * m + i] = f[((size_t)b * C + c) * N + id];
}

// ---------------------------------------------------------------------------
// Ball query: first KNB indices (ascending) within radius; pad with first hit
// ---------------------------------------------------------------------------
__global__ void ball_query(const float* __restrict__ cp, const float* __restrict__ p,
                           int* __restrict__ gidx, int m, int N) {
  const int b = blockIdx.y;
  const int c = blockIdx.x * blockDim.x + threadIdx.x;
  if (c >= m) return;
  const float cx = cp[((size_t)b * m + c) * 3 + 0];
  const float cy = cp[((size_t)b * m + c) * 3 + 1];
  const float cz = cp[((size_t)b * m + c) * 3 + 2];
  int* out = gidx + ((size_t)b * m + c) * KNB;
  int cnt = 0, first = 0;
  for (int j = 0; j < N && cnt < KNB; ++j) {
    float dx = p[((size_t)b * N + j) * 3 + 0] - cx;
    float dy = p[((size_t)b * N + j) * 3 + 1] - cy;
    float dz = p[((size_t)b * N + j) * 3 + 2] - cz;
    if (dx * dx + dy * dy + dz * dz < R2V) {
      if (cnt == 0) first = j;
      out[cnt++] = j;
    }
  }
  for (; cnt < KNB; ++cnt) out[cnt] = first;
}

// ---------------------------------------------------------------------------
// feat (B, I=3+C, m*KNB):  ch<3 -> gp-cp ; else -> fj-cf
// ---------------------------------------------------------------------------
__global__ void build_feat(const float* __restrict__ p, const float* __restrict__ f,
                           const float* __restrict__ cp, const float* __restrict__ cf,
                           const int* __restrict__ gidx, float* __restrict__ feat,
                           int N, int C, int m, int I, size_t total) {
  size_t id = (size_t)blockIdx.x * blockDim.x + threadIdx.x;
  if (id >= total) return;
  const int NN = m * KNB;
  int nn = (int)(id % (size_t)NN);
  int ch = (int)((id / (size_t)NN) % (size_t)I);
  int b  = (int)(id / ((size_t)NN * I));
  int mi = nn >> 5;
  int k  = nn & 31;
  int gi = gidx[((size_t)b * m + mi) * KNB + k];
  float v;
  if (ch < 3) {
    v = p[((size_t)b * N + gi) * 3 + ch] - cp[((size_t)b * m + mi) * 3 + ch];
  } else {
    int cc = ch - 3;
    v = f[((size_t)b * C + cc) * N + gi] - cf[((size_t)b * C + cc) * m + mi];
  }
  feat[id] = v;
}

// ---------------------------------------------------------------------------
// GEMM: Y[b,o,nn] = sum_i W[o,i] * X[b,i,nn]   via v_wmma_f32_16x16x32_f16.
//
// Block = 256 threads (8 waves as 2(O) x 4(NN)); block tile 64(O) x 128(NN).
// Each wave computes a 32x32 patch = 2x2 WMMA accumulators. K step = 32.
//
// Staging is fully affine per thread: the LDS slot is constant (A) or a
// cheap shift function (B); global pointers advance by constant strides
// (16*I for A, 4*NN for B), so the hot path has no per-element 64-bit
// multiplies and no exec-mask divergence. Tail-of-K and pooled-vs-X
// selection are uniform per K-step (splitC is a multiple of 32).
//
// Split-X mode (splitC>0): channel i<splitC comes from Xpool (B,splitC,Mp)
// broadcast over the K neighbors (= concat([pooled, x]) without
// materializing the concat).
// ---------------------------------------------------------------------------
__global__ void __launch_bounds__(256)
gemm_wmma(const float* __restrict__ W, const float* __restrict__ X,
          const float* __restrict__ Xpool, float* __restrict__ Y,
          int O, int I, int NN, int splitC, int Mp) {
  __shared__ __align__(32) uint32_t sA[4 * 256];  // 64x32 f16, 4 row-groups, lane layout
  __shared__ __align__(32) uint32_t sB[8 * 256];  // 32x128 f16, 8 col-groups, lane layout
  const int t    = threadIdx.x;
  const int lane = t & 31;
  const int w    = t >> 5;
  const int wo   = w & 1;    // O half   -> rows [wo*32, wo*32+32)
  const int wn   = w >> 1;   // NN quart -> cols [wn*32, wn*32+32)
  const int b      = blockIdx.z;
  const int oBase  = blockIdx.y * 64;
  const int nnBase = blockIdx.x * 128;
  const int xC     = I - splitC;

  // ---- per-thread constant staging geometry ----
  // A: thread t owns (mRow0 + 16*r, kA) for r = 0..3
  const int mRow0 = t >> 4;             // 0..15
  const int kA    = (t & 15) * 2;       // even k in 0..30
  const int laneA = ((kA >> 4) << 4) | mRow0;
  const int jA    = (kA & 15) >> 1;
  const int aSlot = laneA * 8 + jA;     // + r*256
  // B: thread t owns (k0 + 4*r, nB) for r = 0..7
  const int nB = t & 127;               // column fastest -> coalesced loads
  const int k0 = (t >> 7) * 2;          // 0 or 2
  const int bSlotBase = (nB >> 4) * 256;
  const int nLane     = nB & 15;

  v8f acc00 = {}; v8f acc01 = {}; v8f acc10 = {}; v8f acc11 = {};

  for (int kk = 0; kk < I; kk += 32) {
    const bool fullK   = (kk + 32 <= I);
    const bool usePool = (splitC > 0) && (kk < splitC);   // uniform: splitC % 32 == 0

    // ---- stage A: 64x32 of W as packed k-pair dwords ----
    {
      const float* wr = W + (size_t)(oBase + mRow0) * I + (kk + kA);
      if (fullK) {
        for (int r = 0; r < 4; ++r) {
          sA[aSlot + r * 256] = pack2h(wr[0], wr[1]);
          wr += (size_t)I * 16;
        }
      } else {
        const int gk = kk + kA;
        for (int r = 0; r < 4; ++r) {
          float w0 = (gk < I)     ? wr[0] : 0.0f;
          float w1 = (gk + 1 < I) ? wr[1] : 0.0f;
          sA[aSlot + r * 256] = pack2h(w0, w1);
          wr += (size_t)I * 16;
        }
      }
    }
    // ---- stage B: 32x128 of X as packed k-pair dwords ----
    {
      const size_t col = (size_t)nnBase + nB;
      if (usePool) {
        const float* xp = Xpool + ((size_t)b * splitC + (kk + k0)) * Mp + (col >> 5);
        for (int r = 0; r < 8; ++r) {
          int k = k0 + 4 * r;
          int slot = bSlotBase + (((k >> 4) << 4) | nLane) * 8 + ((k & 15) >> 1);
          sB[slot] = pack2h(xp[0], xp[Mp]);
          xp += (size_t)Mp * 4;
        }
      } else if (fullK) {
        const float* xr = X + ((size_t)b * xC + (kk + k0 - splitC)) * (size_t)NN + col;
        if (kk + 64 <= I)  // hide next K-step's stream behind this step's WMMAs
          __builtin_prefetch(xr + (size_t)NN * 32, 0, 1);
        for (int r = 0; r < 8; ++r) {
          int k = k0 + 4 * r;
          int slot = bSlotBase + (((k >> 4) << 4) | nLane) * 8 + ((k & 15) >> 1);
          sB[slot] = pack2h(xr[0], xr[NN]);
          xr += (size_t)NN * 4;
        }
      } else {
        const float* xr = X + ((size_t)b * xC + (kk + k0 - splitC)) * (size_t)NN + col;
        for (int r = 0; r < 8; ++r) {
          int k = k0 + 4 * r;
          int gi = kk + k;
          float x0 = (gi < I)     ? xr[0]  : 0.0f;
          float x1 = (gi + 1 < I) ? xr[NN] : 0.0f;
          int slot = bSlotBase + (((k >> 4) << 4) | nLane) * 8 + ((k & 15) >> 1);
          sB[slot] = pack2h(x0, x1);
          xr += (size_t)NN * 4;
        }
      }
    }
    __syncthreads();

    const v16h a0 = *reinterpret_cast<const v16h*>(&sA[(wo * 2 + 0) * 256 + lane * 8]);
    const v16h a1 = *reinterpret_cast<const v16h*>(&sA[(wo * 2 + 1) * 256 + lane * 8]);
    const v16h b0 = *reinterpret_cast<const v16h*>(&sB[(wn * 2 + 0) * 256 + lane * 8]);
    const v16h b1 = *reinterpret_cast<const v16h*>(&sB[(wn * 2 + 1) * 256 + lane * 8]);
    acc00 = __builtin_amdgcn_wmma_f32_16x16x32_f16(false, a0, false, b0, (short)0, acc00, false, false);
    acc01 = __builtin_amdgcn_wmma_f32_16x16x32_f16(false, a0, false, b1, (short)0, acc01, false, false);
    acc10 = __builtin_amdgcn_wmma_f32_16x16x32_f16(false, a1, false, b0, (short)0, acc10, false, false);
    acc11 = __builtin_amdgcn_wmma_f32_16x16x32_f16(false, a1, false, b1, (short)0, acc11, false, false);
    __syncthreads();
  }

  // C/D layout: lane n = lane&15, VGPR j holds M = j + (lane>>4)*8
  const int nOut = lane & 15;
  const int mh   = lane >> 4;
  for (int j = 0; j < 8; ++j) {
    int mo = mh * 8 + j;
    size_t row0 = (size_t)b * O + oBase + wo * 32 + mo;
    size_t row1 = row0 + 16;
    size_t cb   = (size_t)nnBase + wn * 32 + nOut;
    Y[row0 * (size_t)NN + cb]      = acc00[j];
    Y[row0 * (size_t)NN + cb + 16] = acc01[j];
    Y[row1 * (size_t)NN + cb]      = acc10[j];
    Y[row1 * (size_t)NN + cb + 16] = acc11[j];
  }
}

// ---------------------------------------------------------------------------
// Instance-norm stats per (b,o) over NN elements (population variance).
// ---------------------------------------------------------------------------
__global__ void norm_stats(const float* __restrict__ x, float* __restrict__ mean,
                           float* __restrict__ rstd, int NN) {
  __shared__ float s1[256];
  __shared__ float s2[256];
  const int bo = blockIdx.x;
  const int t  = threadIdx.x;
  const float* xb = x + (size_t)bo * NN;
  float a = 0.f, q = 0.f;
  for (int i = t; i < NN; i += 256) { float v = xb[i]; a += v; q += v * v; }
  s1[t] = a; s2[t] = q;
  __syncthreads();
  for (int s = 128; s > 0; s >>= 1) {
    if (t < s) { s1[t] += s1[t + s]; s2[t] += s2[t + s]; }
    __syncthreads();
  }
  if (t == 0) {
    float mu  = s1[0] / (float)NN;
    float var = s2[0] / (float)NN - mu * mu;
    mean[bo] = mu;
    rstd[bo] = rsqrtf(var + EPSV);
  }
}

// Squeeze: s[b,o] = mean(relu((x-mu)*rs)) over NN
__global__ void attn_sum(const float* __restrict__ x, const float* __restrict__ mean,
                         const float* __restrict__ rstd, float* __restrict__ s, int NN) {
  __shared__ float s1[256];
  const int bo = blockIdx.x;
  const int t  = threadIdx.x;
  const float mu = mean[bo];
  const float rs = rstd[bo];
  const float* xb = x + (size_t)bo * NN;
  float a = 0.f;
  for (int i = t; i < NN; i += 256) a += fmaxf((xb[i] - mu) * rs, 0.f);
  s1[t] = a;
  __syncthreads();
  for (int ss = 128; ss > 0; ss >>= 1) {
    if (t < ss) s1[t] += s1[t + ss];
    __syncthreads();
  }
  if (t == 0) s[bo] = s1[0] / (float)NN;
}

// Tiny SE MLP per batch: g = sigmoid(relu(s @ w1^T) @ w2^T)
__global__ void attn_mlp(const float* __restrict__ s, const float* __restrict__ w1,
                         const float* __restrict__ w2, float* __restrict__ g,
                         int O, int H) {
  __shared__ float ss[384];
  __shared__ float hh[96];
  const int b = blockIdx.x;
  const int t = threadIdx.x;
  const int T = blockDim.x;
  for (int i = t; i < O; i += T) ss[i] = s[(size_t)b * O + i];
  __syncthreads();
  for (int h = t; h < H; h += T) {
    float a = 0.f;
    for (int i = 0; i < O; ++i) a += w1[(size_t)h * O + i] * ss[i];
    hh[h] = fmaxf(a, 0.f);
  }
  __syncthreads();
  for (int o = t; o < O; o += T) {
    float a = 0.f;
    for (int h = 0; h < H; ++h) a += w2[(size_t)o * H + h] * hh[h];
    g[(size_t)b * O + o] = 1.f / (1.f + __expf(-a));
  }
}

// x = relu((x-mu)*rs) * g  (in place)
__global__ void norm_apply(float* __restrict__ x, const float* __restrict__ mean,
                           const float* __restrict__ rstd, const float* __restrict__ g,
                           int NN, size_t total) {
  size_t id = (size_t)blockIdx.x * blockDim.x + threadIdx.x;
  if (id >= total) return;
  size_t bo = id / (size_t)NN;
  float v = (x[id] - mean[bo]) * rstd[bo];
  x[id] = fmaxf(v, 0.f) * g[bo];
}

// out[b,o,m] = max_k x[b,o,m,k]
__global__ void pool_max(const float* __restrict__ x, float* __restrict__ out,
                         size_t total) {
  size_t id = (size_t)blockIdx.x * blockDim.x + threadIdx.x;
  if (id >= total) return;
  size_t base = id * KNB;
  float mx = x[base];
  for (int k = 1; k < KNB; ++k) mx = fmaxf(mx, x[base + k]);
  out[id] = mx;
}

// ---------------------------------------------------------------------------
// Host-side stage driver
// ---------------------------------------------------------------------------
struct StageBufs {
  int* idx; int* gidx; float* cf; float* feat; float* yA; float* yB; float* pooled;
  float* mean; float* rstd; float* svec; float* g;
};

static void run_stage(const float* p_cur, const float* f_cur, int N, int C, int m,
                      const float* w1, int I1, const float* a1w1, const float* a1w2,
                      const float* w2, int I2, const float* a2w1, const float* a2w2,
                      int O, int H, float* cp_out, float* f_out,
                      const StageBufs& S, hipStream_t stream) {
  const int NN = m * KNB;

  fps_kernel<<<BATCH, 512, 0, stream>>>(p_cur, S.idx, N, m);
  gather_centers<<<dim3((m + 255) / 256, BATCH), 256, 0, stream>>>(
      p_cur, f_cur, S.idx, cp_out, S.cf, N, m, C);
  ball_query<<<dim3((m + 127) / 128, BATCH), 128, 0, stream>>>(
      cp_out, p_cur, S.gidx, m, N);
  {
    size_t total = (size_t)BATCH * I1 * NN;
    build_feat<<<(unsigned)((total + 255) / 256), 256, 0, stream>>>(
        p_cur, f_cur, cp_out, S.cf, S.gidx, S.feat, N, C, m, I1, total);
  }

  // conv block 1 + norm + SE attention
  gemm_wmma<<<dim3(NN / 128, O / 64, BATCH), 256, 0, stream>>>(
      w1, S.feat, nullptr, S.yA, O, I1, NN, 0, 1);
  norm_stats<<<BATCH * O, 256, 0, stream>>>(S.yA, S.mean, S.rstd, NN);
  attn_sum<<<BATCH * O, 256, 0, stream>>>(S.yA, S.mean, S.rstd, S.svec, NN);
  attn_mlp<<<BATCH, 128, 0, stream>>>(S.svec, a1w1, a1w2, S.g, O, H);
  {
    size_t total = (size_t)BATCH * O * NN;
    norm_apply<<<(unsigned)((total + 255) / 256), 256, 0, stream>>>(
        S.yA, S.mean, S.rstd, S.g, NN, total);
  }

  // pooled branch + conv block 2 (split-X GEMM: no concat materialization)
  {
    size_t total = (size_t)BATCH * O * m;
    pool_max<<<(unsigned)((total + 255) / 256), 256, 0, stream>>>(S.yA, S.pooled, total);
  }
  gemm_wmma<<<dim3(NN / 128, O / 64, BATCH), 256, 0, stream>>>(
      w2, S.yA, S.pooled, S.yB, O, I2, NN, /*splitC=*/O, /*Mp=*/m);
  norm_stats<<<BATCH * O, 256, 0, stream>>>(S.yB, S.mean, S.rstd, NN);
  attn_sum<<<BATCH * O, 256, 0, stream>>>(S.yB, S.mean, S.rstd, S.svec, NN);
  attn_mlp<<<BATCH, 128, 0, stream>>>(S.svec, a2w1, a2w2, S.g, O, H);
  {
    size_t total = (size_t)BATCH * O * NN;
    norm_apply<<<(unsigned)((total + 255) / 256), 256, 0, stream>>>(
        S.yB, S.mean, S.rstd, S.g, NN, total);
  }
  {
    size_t total = (size_t)BATCH * O * m;
    pool_max<<<(unsigned)((total + 255) / 256), 256, 0, stream>>>(S.yB, f_out, total);
  }
}

// ---------------------------------------------------------------------------
// Entry point
// ---------------------------------------------------------------------------
extern "C" void kernel_launch(void* const* d_in, const int* in_sizes, int n_in,
                              void* d_out, int out_size, void* d_ws, size_t ws_size,
                              hipStream_t stream) {
  (void)in_sizes; (void)n_in; (void)out_size; (void)ws_size;

  const float* p       = (const float*)d_in[0];
  const float* f       = (const float*)d_in[1];
  const float* s0_w1   = (const float*)d_in[2];
  const float* s0_a1w1 = (const float*)d_in[3];
  const float* s0_a1w2 = (const float*)d_in[4];
  const float* s0_w2   = (const float*)d_in[5];
  const float* s0_a2w1 = (const float*)d_in[6];
  const float* s0_a2w2 = (const float*)d_in[7];
  const float* s1_w1   = (const float*)d_in[8];
  const float* s1_a1w1 = (const float*)d_in[9];
  const float* s1_a1w2 = (const float*)d_in[10];
  const float* s1_w2   = (const float*)d_in[11];
  const float* s1_a2w1 = (const float*)d_in[12];
  const float* s1_a2w2 = (const float*)d_in[13];

  // Output slices (p, cp1, cp2, f, f1, f2) flattened in return order.
  float* out = (float*)d_out;
  float* out_p   = out;                       // 4*8192*3   = 98304
  float* out_cp1 = out + 98304;               // 4*2048*3   = 24576
  float* out_cp2 = out + 98304 + 24576;       // 4*512*3    = 6144
  float* out_f   = out + 129024;              // 4*3*8192   = 98304
  float* out_f1  = out + 227328;              // 4*192*2048 = 1572864
  float* out_f2  = out + 1800192;             // 4*384*512  = 786432

  hipMemcpyAsync(out_p, p, (size_t)98304 * sizeof(float), hipMemcpyDeviceToDevice, stream);
  hipMemcpyAsync(out_f, f, (size_t)98304 * sizeof(float), hipMemcpyDeviceToDevice, stream);

  // Workspace carve-up (256B aligned); buffers sized for the larger stage.
  uint8_t* w = (uint8_t*)d_ws;
  size_t off = 0;
  auto take = [&](size_t bytes) -> void* {
    void* r = w + off;
    off += (bytes + 255) & ~(size_t)255;
    return r;
  };
  int*   idx0   = (int*)take(sizeof(int) * BATCH * 2048);
  int*   idx1   = (int*)take(sizeof(int) * BATCH * 512);
  int*   gidx0  = (int*)take(sizeof(int) * BATCH * 2048 * KNB);
  int*   gidx1  = (int*)take(sizeof(int) * BATCH * 512 * KNB);
  float* cf0    = (float*)take(sizeof(float) * BATCH * 3 * 2048);
  float* cf1    = (float*)take(sizeof(float) * BATCH * 192 * 512);
  float* feat   = (float*)take(sizeof(float) * (size_t)BATCH * 195 * 16384);
  float* yA     = (float*)take(sizeof(float) * (size_t)BATCH * 192 * 65536);
  float* yB     = (float*)take(sizeof(float) * (size_t)BATCH * 192 * 65536);
  float* pooled = (float*)take(sizeof(float) * (size_t)BATCH * 192 * 2048);
  float* meanb  = (float*)take(sizeof(float) * BATCH * 384);
  float* rstdb  = (float*)take(sizeof(float) * BATCH * 384);
  float* svec   = (float*)take(sizeof(float) * BATCH * 384);
  float* gvec   = (float*)take(sizeof(float) * BATCH * 384);

  StageBufs S0{idx0, gidx0, cf0, feat, yA, yB, pooled, meanb, rstdb, svec, gvec};
  run_stage(p, f, /*N=*/8192, /*C=*/3, /*m=*/2048,
            s0_w1, /*I1=*/6, s0_a1w1, s0_a1w2,
            s0_w2, /*I2=*/384, s0_a2w1, s0_a2w2,
            /*O=*/192, /*H=*/48, out_cp1, out_f1, S0, stream);

  StageBufs S1{idx1, gidx1, cf1, feat, yA, yB, pooled, meanb, rstdb, svec, gvec};
  run_stage(out_cp1, out_f1, /*N=*/2048, /*C=*/192, /*m=*/512,
            s1_w1, /*I1=*/195, s1_a1w1, s1_a1w2,
            s1_w2, /*I2=*/768, s1_a2w1, s1_a2w2,
            /*O=*/384, /*H=*/96, out_cp2, out_f2, S1, stream);
}
